// DotProductAttention_292057776923
// MI455X (gfx1250) — matrix operand order repaired and checked by reading, not compiled, not run on hardware
//
#include <hip/hip_runtime.h>
#include <hip/hip_bf16.h>

// ---------------------------------------------------------------------------
// Windowed attention on gfx1250 (MI455X), two-phase:
//   Phase 1 (prepass): fp32 -> bf16 conversion of Q,K (row-major) and V
//                      (transposed per head: VT[b][dim][key]) into d_ws.
//   Phase 2 (flash):   online-softmax attention, all GEMMs via
//                      v_wmma_f32_16x16x32_bf16; K/VT tiles pulled into LDS
//                      with GLOBAL_LOAD_ASYNC_TO_LDS_B128 when available.
// Fallback: single-pass fused kernel if ws_size is too small.
// ---------------------------------------------------------------------------

typedef __attribute__((ext_vector_type(16))) __bf16 v16bf;
typedef __attribute__((ext_vector_type(8)))  __bf16 v8bf;
typedef __attribute__((ext_vector_type(2)))  __bf16 v2bf;
typedef __attribute__((ext_vector_type(8)))  float  v8f;
typedef __attribute__((ext_vector_type(4)))  int    vint4;

#define SEQ     512
#define HDIM    128
#define NHEADS  8
#define NWIN    16
#define KSTR    136   // K-tile LDS row stride (bf16 elems) — breaks bank conflicts
#define VSTR    72    // V^T LDS row stride (bf16 elems)

#if __has_builtin(__builtin_amdgcn_global_load_async_to_lds_b128)
#define USE_ASYNC_LDS 1
#endif

// addrspace-qualified pointer types matching the async-LDS builtin signature
typedef __attribute__((address_space(1))) vint4* as1_v4i_p;
typedef __attribute__((address_space(3))) vint4* as3_v4i_p;

__device__ __forceinline__ unsigned short bfbits(float x) {
  unsigned u = __float_as_uint(x);
  u += 0x7FFFu + ((u >> 16) & 1u);      // round-to-nearest-even
  return (unsigned short)(u >> 16);
}

__device__ __forceinline__ unsigned bf2pack(float a, float b) {
#if __has_builtin(__builtin_amdgcn_cvt_pk_bf16_f32)
  v2bf r = __builtin_amdgcn_cvt_pk_bf16_f32(a, b);
  return __builtin_bit_cast(unsigned, r);
#else
  return (unsigned)bfbits(a) | ((unsigned)bfbits(b) << 16);
#endif
}

// 16-byte global -> LDS copy; async DMA path when the toolchain exposes it.
__device__ __forceinline__ void cp16_g2l(unsigned short* l, const unsigned short* g) {
#ifdef USE_ASYNC_LDS
  __builtin_amdgcn_global_load_async_to_lds_b128((as1_v4i_p)g, (as3_v4i_p)l, 0, 0);
#else
  *(uint4*)l = *(const uint4*)g;
#endif
}

__device__ __forceinline__ void async_drain() {
#ifdef USE_ASYNC_LDS
#if __has_builtin(__builtin_amdgcn_s_wait_asynccnt)
  __builtin_amdgcn_s_wait_asynccnt(0);
#else
  asm volatile("s_wait_asynccnt 0x0" ::: "memory");
#endif
#endif
}

__device__ __forceinline__ float redmax16(float x) {
  x = fmaxf(x, __shfl_xor(x, 1, 32));
  x = fmaxf(x, __shfl_xor(x, 2, 32));
  x = fmaxf(x, __shfl_xor(x, 4, 32));
  x = fmaxf(x, __shfl_xor(x, 8, 32));
  return x;
}
__device__ __forceinline__ float redsum16(float x) {
  x += __shfl_xor(x, 1, 32);
  x += __shfl_xor(x, 2, 32);
  x += __shfl_xor(x, 4, 32);
  x += __shfl_xor(x, 8, 32);
  return x;
}

__device__ __forceinline__ v16bf cat8(v8bf lo, v8bf hi) {
  return __builtin_shufflevector(lo, hi, 0,1,2,3,4,5,6,7,8,9,10,11,12,13,14,15);
}

// ============================ Phase-1 kernels ==============================

// Elementwise fp32 -> bf16 (8 elems/thread, grid-stride).
__global__ __launch_bounds__(256)
void cvt_bf16_kernel(const float* __restrict__ src,
                     unsigned short* __restrict__ dst, int n8) {
  const int stride = gridDim.x * blockDim.x;
  for (int i = blockIdx.x * blockDim.x + threadIdx.x; i < n8; i += stride) {
    const float4 f0 = *(const float4*)(src + 8 * (size_t)i);
    const float4 f1 = *(const float4*)(src + 8 * (size_t)i + 4);
    uint4 o;
    o.x = bf2pack(f0.x, f0.y);
    o.y = bf2pack(f0.z, f0.w);
    o.z = bf2pack(f1.x, f1.y);
    o.w = bf2pack(f1.z, f1.w);
    *(uint4*)(dst + 8 * (size_t)i) = o;
  }
}

// V[b][key][dim] fp32 -> VT[b][dim][key] bf16, 64-key tile per block via LDS.
__global__ __launch_bounds__(128)
void vT_bf16_kernel(const float* __restrict__ V, unsigned short* __restrict__ VT) {
  const int b  = blockIdx.x;
  const int kb = blockIdx.y * 64;
  const int tid = threadIdx.x;
  __shared__ __align__(16) unsigned short Vl[64 * KSTR];

  const int r  = tid >> 1;
  const int dh = (tid & 1) * 64;
  const float* vg = V + ((size_t)b * SEQ + kb + r) * HDIM + dh;
#pragma unroll
  for (int i = 0; i < 16; ++i) {
    const float4 f = *(const float4*)(vg + 4 * i);
    uint2 p;
    p.x = bf2pack(f.x, f.y);
    p.y = bf2pack(f.z, f.w);
    *(uint2*)&Vl[r * KSTR + dh + 4 * i] = p;
  }
  __syncthreads();

  // thread t gathers dim-row t (64 keys) and writes it contiguously
  unsigned short* outp = VT + ((size_t)b * HDIM + tid) * SEQ + kb;
#pragma unroll
  for (int kk = 0; kk < 64; kk += 8) {
    uint4 o;
    o.x = (unsigned)Vl[(kk + 0) * KSTR + tid] | ((unsigned)Vl[(kk + 1) * KSTR + tid] << 16);
    o.y = (unsigned)Vl[(kk + 2) * KSTR + tid] | ((unsigned)Vl[(kk + 3) * KSTR + tid] << 16);
    o.z = (unsigned)Vl[(kk + 4) * KSTR + tid] | ((unsigned)Vl[(kk + 5) * KSTR + tid] << 16);
    o.w = (unsigned)Vl[(kk + 6) * KSTR + tid] | ((unsigned)Vl[(kk + 7) * KSTR + tid] << 16);
    *(uint4*)(outp + kk) = o;
  }
}

// ============================ Phase-2 kernel ===============================

__global__ __launch_bounds__(128)
void fa_bf16_kernel(const unsigned short* __restrict__ Qb,
                    const unsigned short* __restrict__ Kb,
                    const unsigned short* __restrict__ VT,
                    const int*   __restrict__ vlen,
                    const float* __restrict__ WM,
                    float*       __restrict__ Out) {
  const int b    = blockIdx.x;
  const int qb   = blockIdx.y * 64;
  const int tid  = threadIdx.x;
  const int lane = tid & 31;
  const int wave = tid >> 5;
  const int half = lane >> 4;
  const int ln   = lane & 15;

  const int w  = (b / NHEADS) % NWIN;
  const int vl = vlen[b];

  __shared__ __align__(16) unsigned short Kl[64 * KSTR];
  __shared__ __align__(16) unsigned short Vt[HDIM * VSTR];
  __shared__ __align__(16) unsigned short Pl[4 * 16 * 64];

  const int kend0 = (vl > 0) ? vl : SEQ;
  const int kend  = ((kend0 + 63) >> 6) << 6;

  // Q A-fragments straight from bf16 global (two contiguous 16B loads each)
  const int qrowA = qb + wave * 16 + ln;
  const unsigned short* qp = Qb + ((size_t)b * SEQ + qrowA) * HDIM;
  v16bf qa[4];
#pragma unroll
  for (int j = 0; j < 4; ++j) {
    const int k0 = 32 * j + 8 * half;
    qa[j] = cat8(*(const v8bf*)(qp + k0), *(const v8bf*)(qp + k0 + 16));
  }

  v8f o[8] = {};
  float mrow[8], lrow[8];
#pragma unroll
  for (int v = 0; v < 8; ++v) { mrow[v] = -3e38f; lrow[v] = 0.0f; }

  const float scale = 0.08838834764831843f;   // 1/sqrt(128)
  unsigned short* pw = &Pl[wave * 1024];

  const unsigned short* kg0 = Kb + (size_t)b * SEQ * HDIM;
  const unsigned short* vg0 = VT + (size_t)b * HDIM * SEQ;

  for (int kb = 0; kb < kend; kb += 64) {
    __syncthreads();    // previous tile fully consumed by all waves

    // ---- Tile fill: 16B/lane copies, async-to-LDS when available ----
    {
      const unsigned short* kgb = kg0 + (size_t)(kb) * HDIM;  // [key][dim]
      const int r   = tid >> 1;
      const int ch0 = (tid & 1) * 8;
#pragma unroll
      for (int i = 0; i < 8; ++i) {
        const int c = ch0 + i;
        cp16_g2l(&Kl[r * KSTR + c * 8], kgb + (size_t)r * HDIM + c * 8);
      }
#pragma unroll
      for (int i = 0; i < 8; ++i)
        cp16_g2l(&Vt[tid * VSTR + i * 8], vg0 + (size_t)tid * SEQ + kb + i * 8);
    }
    async_drain();
    __syncthreads();

    // ---- S = Q K^T : 4 key tiles x 4 k-chunks of 32 ----
    v8f st[4];
#pragma unroll
    for (int t = 0; t < 4; ++t) {
      v8f acc = {};
#pragma unroll
      for (int j = 0; j < 4; ++j) {
        const int off = (16 * t + ln) * KSTR + 32 * j + 8 * half;
        acc = __builtin_amdgcn_wmma_f32_16x16x32_bf16(
            false, qa[j], false,
            cat8(*(const v8bf*)&Kl[off], *(const v8bf*)&Kl[off + 16]),
            (short)0, acc, false, false);
      }
      st[t] = acc;
    }

    // ---- scale + window mask + valid-len mask; online softmax ----
    float mnew[8];
#pragma unroll
    for (int v = 0; v < 8; ++v) {
      const int qr = qb + wave * 16 + v + 8 * half;
      const float* wmp = WM + ((size_t)w * SEQ + qr) * SEQ + kb + ln;
      float mx = -3e38f;
#pragma unroll
      for (int t = 0; t < 4; ++t) {
        const int key = kb + 16 * t + ln;
        float s = st[t][v] * scale + wmp[16 * t];
        if (key >= vl) s = -1e6f;
        st[t][v] = s;
        mx = fmaxf(mx, s);
      }
      mx      = redmax16(mx);
      mnew[v] = fmaxf(mrow[v], mx);
    }

#pragma unroll
    for (int v = 0; v < 8; ++v) {
      const float alpha = __expf(mrow[v] - mnew[v]);
      float rs = 0.0f;
#pragma unroll
      for (int t = 0; t < 4; ++t) {
        const float p = __expf(st[t][v] - mnew[v]);
        st[t][v] = p;
        rs += p;
      }
      rs      = redsum16(rs);
      lrow[v] = lrow[v] * alpha + rs;
      mrow[v] = mnew[v];
#pragma unroll
      for (int nt = 0; nt < 8; ++nt) o[nt][v] *= alpha;
    }

    // ---- Stage P (C layout -> per-wave LDS strip, bf16) ----
#pragma unroll
    for (int t = 0; t < 4; ++t)
#pragma unroll
      for (int v = 0; v < 8; ++v)
        pw[(v + 8 * half) * 64 + 16 * t + ln] = bfbits(st[t][v]);

    // ---- O += P * V ----
#pragma unroll
    for (int kc = 0; kc < 64; kc += 32) {
      const int aoff = ln * 64 + kc + 8 * half;
      const v16bf af = cat8(*(const v8bf*)&pw[aoff], *(const v8bf*)&pw[aoff + 16]);
#pragma unroll
      for (int nt = 0; nt < 8; ++nt) {
        const int boff = (nt * 16 + ln) * VSTR + kc + 8 * half;
        o[nt] = __builtin_amdgcn_wmma_f32_16x16x32_bf16(
            false, af, false,
            cat8(*(const v8bf*)&Vt[boff], *(const v8bf*)&Vt[boff + 16]),
            (short)0, o[nt], false, false);
      }
    }
  }

  float inv[8];
#pragma unroll
  for (int v = 0; v < 8; ++v) inv[v] = 1.0f / lrow[v];
#pragma unroll
  for (int nt = 0; nt < 8; ++nt)
#pragma unroll
    for (int v = 0; v < 8; ++v) {
      const int qr = qb + wave * 16 + v + 8 * half;
      Out[((size_t)b * SEQ + qr) * HDIM + nt * 16 + ln] = o[nt][v] * inv[v];
    }
}

// ================= Fallback: fused single-pass kernel ======================

__global__ __launch_bounds__(128)
void fa_windowed_kernel(const float* __restrict__ Q,
                        const float* __restrict__ Kp,
                        const float* __restrict__ Vp,
                        const int*   __restrict__ vlen,
                        const float* __restrict__ WM,
                        float*       __restrict__ Out) {
  const int b    = blockIdx.x;
  const int qb   = blockIdx.y * 64;
  const int tid  = threadIdx.x;
  const int lane = tid & 31;
  const int wave = tid >> 5;
  const int half = lane >> 4;
  const int ln   = lane & 15;

  const int w  = (b / NHEADS) % NWIN;
  const int vl = vlen[b];

  __shared__ __align__(16) unsigned short Kl[64 * KSTR];
  __shared__ __align__(16) unsigned short Vt[HDIM * VSTR];
  __shared__ __align__(16) unsigned short Pl[4 * 16 * 64];

  const int kend0 = (vl > 0) ? vl : SEQ;
  const int kend  = ((kend0 + 63) >> 6) << 6;

  const int qrowA = qb + wave * 16 + ln;
  const float* qp = Q + ((size_t)b * SEQ + qrowA) * HDIM;
  v16bf qa[4];
#pragma unroll
  for (int j = 0; j < 4; ++j) {
    union { unsigned short u[16]; v16bf v; } tmp;
    const int k0 = 32 * j + 8 * half;
#pragma unroll
    for (int e = 0; e < 8; ++e) tmp.u[e]     = bfbits(qp[k0 + e]);
#pragma unroll
    for (int e = 0; e < 8; ++e) tmp.u[8 + e] = bfbits(qp[k0 + 16 + e]);
    qa[j] = tmp.v;
  }

  v8f o[8] = {};
  float mrow[8], lrow[8];
#pragma unroll
  for (int v = 0; v < 8; ++v) { mrow[v] = -3e38f; lrow[v] = 0.0f; }

  const float scale = 0.08838834764831843f;
  unsigned short* pw = &Pl[wave * 1024];

  for (int kb = 0; kb < kend; kb += 64) {
    __syncthreads();
    {
      const int row = tid >> 1;
      const int dh  = (tid & 1) * 64;
      const float* kg = Kp + ((size_t)b * SEQ + kb + row) * HDIM + dh;
      const float* vg = Vp + ((size_t)b * SEQ + kb + row) * HDIM + dh;
#pragma unroll
      for (int i = 0; i < 16; ++i) {
        const float4 fk = *(const float4*)(kg + 4 * i);
        const int db = dh + 4 * i;
        uint2 pk;
        pk.x = bf2pack(fk.x, fk.y);
        pk.y = bf2pack(fk.z, fk.w);
        *(uint2*)&Kl[row * KSTR + db] = pk;
        const float4 fv = *(const float4*)(vg + 4 * i);
        Vt[(db + 0) * VSTR + row] = bfbits(fv.x);
        Vt[(db + 1) * VSTR + row] = bfbits(fv.y);
        Vt[(db + 2) * VSTR + row] = bfbits(fv.z);
        Vt[(db + 3) * VSTR + row] = bfbits(fv.w);
      }
    }
    __syncthreads();

    v8f st[4];
#pragma unroll
    for (int t = 0; t < 4; ++t) {
      v8f acc = {};
#pragma unroll
      for (int j = 0; j < 4; ++j) {
        const int off = (16 * t + ln) * KSTR + 32 * j + 8 * half;
        acc = __builtin_amdgcn_wmma_f32_16x16x32_bf16(
            false, qa[j], false,
            cat8(*(const v8bf*)&Kl[off], *(const v8bf*)&Kl[off + 16]),
            (short)0, acc, false, false);
      }
      st[t] = acc;
    }

    float mnew[8];
#pragma unroll
    for (int v = 0; v < 8; ++v) {
      const int qr = qb + wave * 16 + v + 8 * half;
      const float* wmp = WM + ((size_t)w * SEQ + qr) * SEQ + kb + ln;
      float mx = -3e38f;
#pragma unroll
      for (int t = 0; t < 4; ++t) {
        const int key = kb + 16 * t + ln;
        float s = st[t][v] * scale + wmp[16 * t];
        if (key >= vl) s = -1e6f;
        st[t][v] = s;
        mx = fmaxf(mx, s);
      }
      mx      = redmax16(mx);
      mnew[v] = fmaxf(mrow[v], mx);
    }

#pragma unroll
    for (int v = 0; v < 8; ++v) {
      const float alpha = __expf(mrow[v] - mnew[v]);
      float rs = 0.0f;
#pragma unroll
      for (int t = 0; t < 4; ++t) {
        const float p = __expf(st[t][v] - mnew[v]);
        st[t][v] = p;
        rs += p;
      }
      rs      = redsum16(rs);
      lrow[v] = lrow[v] * alpha + rs;
      mrow[v] = mnew[v];
#pragma unroll
      for (int nt = 0; nt < 8; ++nt) o[nt][v] *= alpha;
    }

#pragma unroll
    for (int t = 0; t < 4; ++t)
#pragma unroll
      for (int v = 0; v < 8; ++v)
        pw[(v + 8 * half) * 64 + 16 * t + ln] = bfbits(st[t][v]);

#pragma unroll
    for (int kc = 0; kc < 64; kc += 32) {
      const int aoff = ln * 64 + kc + 8 * half;
      const v16bf af = cat8(*(const v8bf*)&pw[aoff], *(const v8bf*)&pw[aoff + 16]);
#pragma unroll
      for (int nt = 0; nt < 8; ++nt) {
        const int boff = (nt * 16 + ln) * VSTR + kc + 8 * half;
        o[nt] = __builtin_amdgcn_wmma_f32_16x16x32_bf16(
            false, af, false,
            cat8(*(const v8bf*)&Vt[boff], *(const v8bf*)&Vt[boff + 16]),
            (short)0, o[nt], false, false);
      }
    }
  }

  float inv[8];
#pragma unroll
  for (int v = 0; v < 8; ++v) inv[v] = 1.0f / lrow[v];
#pragma unroll
  for (int nt = 0; nt < 8; ++nt)
#pragma unroll
    for (int v = 0; v < 8; ++v) {
      const int qr = qb + wave * 16 + v + 8 * half;
      Out[((size_t)b * SEQ + qr) * HDIM + nt * 16 + ln] = o[nt][v] * inv[v];
    }
}

// ================================ Host =====================================

extern "C" void kernel_launch(void* const* d_in, const int* in_sizes, int n_in,
                              void* d_out, int out_size, void* d_ws, size_t ws_size,
                              hipStream_t stream) {
  (void)n_in; (void)out_size;
  const float* Q  = (const float*)d_in[0];
  const float* K  = (const float*)d_in[1];
  const float* V  = (const float*)d_in[2];
  const int*   vl = (const int*)d_in[3];
  const float* wm = (const float*)d_in[4];
  float* out = (float*)d_out;

  const int n = in_sizes[3];                     // 256 batch-heads
  const size_t nel = (size_t)in_sizes[0];        // 256*512*128 elements
  const size_t need = 3 * nel * sizeof(unsigned short);

  dim3 grid(n, SEQ / 64, 1);

  if (ws_size >= need) {
    unsigned short* Qb = (unsigned short*)d_ws;
    unsigned short* Kb = Qb + nel;
    unsigned short* VT = Kb + nel;
    const int n8 = (int)(nel / 8);
    cvt_bf16_kernel<<<2048, 256, 0, stream>>>(Q, Qb, n8);
    cvt_bf16_kernel<<<2048, 256, 0, stream>>>(K, Kb, n8);
    vT_bf16_kernel<<<grid, 128, 0, stream>>>(V, VT);
    fa_bf16_kernel<<<grid, 128, 0, stream>>>(Qb, Kb, VT, vl, wm, out);
  } else {
    fa_windowed_kernel<<<grid, 128, 0, stream>>>(Q, K, V, vl, wm, out);
  }
}
